// SelfAttentionV2_37314675867764
// MI455X (gfx1250) — matrix operand order repaired and checked by reading, not compiled, not run on hardware
//
#include <hip/hip_runtime.h>

typedef __bf16 bf16;
typedef __attribute__((ext_vector_type(8)))  __bf16 v8bf;
typedef __attribute__((ext_vector_type(16))) __bf16 v16bf;
typedef __attribute__((ext_vector_type(8)))  float  v8f;
typedef __attribute__((ext_vector_type(4)))  unsigned int u32x4;
typedef __attribute__((ext_vector_type(8)))  int i32x8;
typedef __attribute__((ext_vector_type(4)))  int i32x4;

#define DIM 256
#define SEQ 4096
#define NB  4
// 1/sqrt(256) * log2(e): fold softmax scale + exp->exp2 conversion into Q
#define QSCALE 0.09016844005556021f

static __device__ __forceinline__ v16bf cat16(v8bf lo, v8bf hi) {
  return __builtin_shufflevector(lo, hi, 0,1,2,3,4,5,6,7,8,9,10,11,12,13,14,15);
}
// A-fragment (16x32 bf16): per-lane chunks at +0..7 and +16..23 halves
static __device__ __forceinline__ v16bf load_afrag(const bf16* p) {
  return cat16(*(const v8bf*)p, *(const v8bf*)(p + 16));
}
// B-fragment (32x16 bf16): per-lane chunks at +0..7 and +16..23 halves
static __device__ __forceinline__ v16bf load_bfrag(const bf16* p) {
  return cat16(*(const v8bf*)p, *(const v8bf*)(p + 16));
}
static __device__ __forceinline__ v8f wmma_bf16(v16bf a, v16bf b, v8f c) {
  return __builtin_amdgcn_wmma_f32_16x16x32_bf16(false, a, false, b, (short)0, c,
                                                 false, false);
}
static __device__ __forceinline__ float rmax16(float v) {
  v = fmaxf(v, __shfl_xor(v, 1, 32));
  v = fmaxf(v, __shfl_xor(v, 2, 32));
  v = fmaxf(v, __shfl_xor(v, 4, 32));
  v = fmaxf(v, __shfl_xor(v, 8, 32));
  return v;
}
static __device__ __forceinline__ float rsum16(float v) {
  v += __shfl_xor(v, 1, 32);
  v += __shfl_xor(v, 2, 32);
  v += __shfl_xor(v, 4, 32);
  v += __shfl_xor(v, 8, 32);
  return v;
}

// LDS byte offset of a __shared__ object (generic -> AS(3) -> int)
static __device__ __forceinline__ unsigned int lds_addr_of(const void* p) {
  return (unsigned int)(size_t)(const __attribute__((address_space(3))) void*)p;
}

// Tensor Data Mover: 2D tile load, bf16 elements, LDS row padding.
// D# per CDNA5 ISA ch.8: group0 = {flags, lds_addr, global_addr, type=2},
// group1 = {mode bits, tensor dims/strides, tile dims}, groups 2/3 zero (<=2D).
static __device__ __forceinline__ void tdm_load_2d(
    unsigned int lds_addr, const void* gptr,
    unsigned int dim0, unsigned int dim1, unsigned int stride0,
    unsigned int tile0, unsigned int tile1,
    unsigned int pad_interval, unsigned int pad_amount) {
  unsigned long long ga = (unsigned long long)(size_t)gptr;
  u32x4 g0;
  g0[0] = 1u;                                   // count=1, user mode
  g0[1] = lds_addr;                             // LDS byte address
  g0[2] = (unsigned int)ga;                     // global_addr[31:0]
  g0[3] = (unsigned int)((ga >> 32) & 0x1FFFFFFu) | (2u << 30);  // type=2
  i32x8 g1;
  g1[0] = (int)((1u << 16)                      // data_size = 2 bytes
              | (1u << 20)                      // pad_enable
              | (pad_interval << 22)
              | (pad_amount << 25));
  g1[1] = (int)(dim0 << 16);                    // tensor_dim0[15:0]
  g1[2] = (int)((dim0 >> 16) | (dim1 << 16));   // dim0[31:16] | dim1[15:0]
  g1[3] = (int)((dim1 >> 16) | (tile0 << 16));  // dim1[31:16] | tile_dim0
  g1[4] = (int)(tile1 & 0xFFFFu);               // tile_dim1 (tile_dim2=0)
  g1[5] = (int)stride0;                         // tensor_dim0_stride[31:0]
  g1[6] = 0;                                    // stride0[47:32] | stride1 lo
  g1[7] = 0;
  i32x4 z4 = {0, 0, 0, 0};
  i32x8 z8 = {0, 0, 0, 0, 0, 0, 0, 0};
  // 6-arg toolchain variant: (g0, g1, g2, g3, extra, cpol)
  __builtin_amdgcn_tensor_load_to_lds(g0, g1, z4, z4, z8, 0);
}

// --- Kernel 0: W [256][768] f32 -> Wt [768][256] bf16 (transposed) ---
__global__ void __launch_bounds__(256) k_wtrans(const float* __restrict__ W,
                                                bf16* __restrict__ Wt) {
  int n = blockIdx.x;   // 0..767
  int k = threadIdx.x;  // 0..255
  Wt[(size_t)n * DIM + k] = (bf16)W[(size_t)k * (3 * DIM) + n];
}

// --- Kernel 1: QKV = X @ W + b ; write Qb (scaled), Kb, Vt(b,d,s) in bf16 ---
__global__ void __launch_bounds__(256) k_qkv(const float* __restrict__ X,
                                             const float* __restrict__ bias,
                                             const bf16* __restrict__ Wt,
                                             bf16* __restrict__ Qb,
                                             bf16* __restrict__ Kb,
                                             bf16* __restrict__ Vt) {
  const int lane = threadIdx.x & 31;
  const int w    = threadIdx.x >> 5;   // wave 0..7 -> n-tiles w*6..w*6+5
  const int hi   = lane >> 4;
  const int lp   = lane & 15;
  const int m0   = blockIdx.x << 4;    // 16-row M tile

  v8f acc[6] = {};
  const float* xr = X + (size_t)(m0 + lp) * DIM + (hi ? 8 : 0);

  #pragma unroll
  for (int kk = 0; kk < 8; ++kk) {
    const float* xp = xr + kk * 32;
    v16bf a;
    #pragma unroll
    for (int i = 0; i < 8; ++i) a[i] = (bf16)xp[i];
    #pragma unroll
    for (int i = 0; i < 8; ++i) a[8 + i] = (bf16)xp[16 + i];
    const int ko = kk * 32 + (hi ? 8 : 0);
    #pragma unroll
    for (int t = 0; t < 6; ++t) {
      const int ng = w * 6 + t;
      const bf16* wp = Wt + (size_t)(ng * 16 + lp) * DIM + ko;
      acc[t] = wmma_bf16(a, load_bfrag(wp), acc[t]);
    }
  }

  #pragma unroll
  for (int t = 0; t < 6; ++t) {
    const int ng    = w * 6 + t;
    const int which = ng >> 4;                 // 0=Q 1=K 2=V
    const int nc    = ((ng & 15) << 4) + lp;   // column within 256
    const float bv  = bias[(which << 8) + nc];
    #pragma unroll
    for (int j = 0; j < 8; ++j) {
      const int m   = m0 + j + hi * 8;         // global row (b*4096+s)
      const float v = acc[t][j] + bv;
      if (which == 0)
        Qb[(size_t)m * DIM + nc] = (bf16)(v * QSCALE);
      else if (which == 1)
        Kb[(size_t)m * DIM + nc] = (bf16)v;
      else
        Vt[((size_t)((m >> 12) * DIM + nc)) * SEQ + (m & 4095)] = (bf16)v;
    }
  }
}

// --- Kernel 2: flash attention. 4 waves/block; wave owns 16 query rows.
//     K/V tiles staged by the Tensor Data Mover, double buffered. ---
__global__ void __launch_bounds__(128) k_flash(const bf16* __restrict__ Qb,
                                               const bf16* __restrict__ Kb,
                                               const bf16* __restrict__ Vt,
                                               float* __restrict__ out) {
  // padded rows (TDM pad_enable produces the same strides) -> conflict-free
  __shared__ bf16 Kt[2][32][264];    // key tiles, row-major   (2 x 16.9 KB)
  __shared__ bf16 Vs[2][256][40];    // V tiles, d-major       (2 x 20.5 KB)
  __shared__ bf16 Ps[4][16][40];     // per-wave P transpose   (    5.1 KB)

  const int tid  = threadIdx.x;
  const int lane = tid & 31;
  const int w    = tid >> 5;         // 0..3
  const int hi   = lane >> 4;
  const int lp   = lane & 15;
  const int bb   = blockIdx.x >> 6;
  const int q0   = (blockIdx.x & 63) << 6;

  const int   qrow   = q0 + w * 16 + lp;
  const bf16* qbase  = Qb + (size_t)(bb * SEQ + qrow) * DIM + (hi ? 8 : 0);
  const bf16* kbase0 = Kb + (size_t)(bb * SEQ) * DIM;
  const bf16* vbase0 = Vt + (size_t)bb * DIM * SEQ;

  v8f o[16] = {};
  float mrow[8], lrow[8];
  #pragma unroll
  for (int j = 0; j < 8; ++j) { mrow[j] = -3.0e38f; lrow[j] = 0.f; }

  // prefetch tile 0 via TDM (wave 0: K, wave 1: V)
  if (w == 0)
    tdm_load_2d(lds_addr_of(&Kt[0][0][0]), kbase0,
                DIM, SEQ, DIM, /*tile*/ DIM, 32, /*pad*/ 6, 3);   // 512B+16B rows
  else if (w == 1)
    tdm_load_2d(lds_addr_of(&Vs[0][0][0]), vbase0,
                SEQ, DIM, SEQ, /*tile*/ 32, DIM, /*pad*/ 3, 3);   // 64B+16B rows

  const int NIT = SEQ / 32;
  for (int it = 0; it < NIT; ++it) {
    const int cur = it & 1;
    const int nxt = cur ^ 1;
    const int j1  = (it + 1) * 32;

    if (it + 1 < NIT) {   // kick next tile's DMA, then wait for current
      if (w == 0)
        tdm_load_2d(lds_addr_of(&Kt[nxt][0][0]), kbase0 + (size_t)j1 * DIM,
                    DIM, SEQ, DIM, DIM, 32, 6, 3);
      else if (w == 1)
        tdm_load_2d(lds_addr_of(&Vs[nxt][0][0]), vbase0 + j1,
                    SEQ, DIM, SEQ, 32, DIM, 3, 3);
      if (w < 2) __builtin_amdgcn_s_wait_tensorcnt(1);
    } else {
      if (w < 2) __builtin_amdgcn_s_wait_tensorcnt(0);
    }
    __syncthreads();   // tile `cur` visible to all waves

    // S = Q . K^T  (two 16x16 tiles covering 32 keys)
    v8f s0 = {}, s1 = {};
    #pragma unroll
    for (int kk = 0; kk < 8; ++kk) {
      v16bf a = load_afrag(qbase + kk * 32);
      const int ko = kk * 32 + (hi ? 8 : 0);
      v16bf b0 = load_bfrag(&Kt[cur][lp][ko]);
      v16bf b1 = load_bfrag(&Kt[cur][16 + lp][ko]);
      s0 = wmma_bf16(a, b0, s0);
      s1 = wmma_bf16(a, b1, s1);
    }

    // online softmax (log2 domain; scale folded into Q)
    float alpha[8];
    bf16* pr = &Ps[w][0][0];
    #pragma unroll
    for (int j = 0; j < 8; ++j) {
      float v  = rmax16(fmaxf(s0[j], s1[j]));
      float mn = fmaxf(mrow[j], v);
      float a_ = exp2f(mrow[j] - mn);
      float e0 = exp2f(s0[j] - mn);
      float e1 = exp2f(s1[j] - mn);
      lrow[j]  = lrow[j] * a_ + rsum16(e0 + e1);
      mrow[j]  = mn;
      alpha[j] = a_;
      const int r = j + hi * 8;
      pr[r * 40 + lp]      = (bf16)e0;
      pr[r * 40 + 16 + lp] = (bf16)e1;
    }
    #pragma unroll
    for (int dt = 0; dt < 16; ++dt)
      #pragma unroll
      for (int j = 0; j < 8; ++j) o[dt][j] *= alpha[j];

    // P was written in D-layout; reload as A-fragment (wave-local region)
    asm volatile("s_wait_dscnt 0" ::: "memory");
    const int c0 = hi ? 8 : 0;
    v16bf pf = cat16(*(const v8bf*)&pr[lp * 40 + c0],
                     *(const v8bf*)&pr[lp * 40 + c0 + 16]);

    // O += P . V
    const int vo = hi ? 8 : 0;
    #pragma unroll
    for (int dt = 0; dt < 16; ++dt) {
      v16bf bv = load_bfrag(&Vs[cur][dt * 16 + lp][vo]);
      o[dt] = wmma_bf16(pf, bv, o[dt]);
    }
    __syncthreads();   // all waves done with tile `cur` before its reuse
  }

  // normalize + store fp32
  float rl[8];
  #pragma unroll
  for (int j = 0; j < 8; ++j) rl[j] = 1.0f / lrow[j];
  float* ob = out + (size_t)(bb * SEQ + q0 + w * 16) * DIM;
  #pragma unroll
  for (int dt = 0; dt < 16; ++dt)
    #pragma unroll
    for (int j = 0; j < 8; ++j)
      ob[(size_t)(j + hi * 8) * DIM + dt * 16 + lp] = o[dt][j] * rl[j];
}

extern "C" void kernel_launch(void* const* d_in, const int* in_sizes, int n_in,
                              void* d_out, int out_size, void* d_ws, size_t ws_size,
                              hipStream_t stream) {
  (void)in_sizes; (void)n_in; (void)out_size; (void)ws_size;
  const float* X    = (const float*)d_in[0];
  const float* W    = (const float*)d_in[1];
  const float* bias = (const float*)d_in[2];
  float* out = (float*)d_out;

  char* ws = (char*)d_ws;
  bf16* Wt = (bf16*)ws;                              // 768*256*2   = 384 KB
  bf16* Qb = (bf16*)(ws + (size_t)768 * 256 * 2);    // 16384*256*2 = 8 MB
  bf16* Kb = Qb + (size_t)NB * SEQ * DIM;            // 8 MB
  bf16* Vt = Kb + (size_t)NB * SEQ * DIM;            // 8 MB (layout [b][d][s])

  k_wtrans<<<dim3(3 * DIM), dim3(256), 0, stream>>>(W, Wt);
  k_qkv<<<dim3(NB * SEQ / 16), dim3(256), 0, stream>>>(X, bias, Wt, Qb, Kb, Vt);
  k_flash<<<dim3(NB * SEQ / 64), dim3(128), 0, stream>>>(Qb, Kb, Vt, out);
}
